// QAEVABlock_3985729651529
// MI455X (gfx1250) — matrix operand order repaired and checked by reading, not compiled, not run on hardware
//
#include <hip/hip_runtime.h>
#include <hip/hip_bf16.h>
#include <math.h>

#define DEV __device__ __forceinline__

// ---------------- problem constants ----------------
#define BB 4
#define NN 1024
#define CC 1024
#define HH 16
#define HD 64
#define II 77
#define IDD 768
#define HID 4096
#define SS 1101          // I + N
#define SP 1120          // padded key stride for V^T (mult of 16, >= 1088+32)

typedef __attribute__((ext_vector_type(16))) __bf16 v16bf;
typedef __attribute__((ext_vector_type(8)))  float  v8f;

union AFrag { uint4 u[2]; v16bf v; };

DEV unsigned short f2bf(float f) {
  unsigned int u = __float_as_uint(f);
  unsigned int r = u + 0x7FFFu + ((u >> 16) & 1u);   // RNE
  return (unsigned short)(r >> 16);
}
DEV float gelu_exact(float x) {
  return 0.5f * x * (1.0f + erff(x * 0.7071067811865475f));
}
DEV v8f vzero8() { v8f z = {0.f,0.f,0.f,0.f,0.f,0.f,0.f,0.f}; return z; }
DEV uint4 uz4() { return make_uint4(0u,0u,0u,0u); }
DEV v8f wmma_bf16(AFrag a, AFrag b, v8f c) {
  return __builtin_amdgcn_wmma_f32_16x16x32_bf16(false, a.v, false, b.v,
                                                 (short)0, c, false, false);
}

// ---------------- elementwise kernels ----------------
__global__ void cvt_bf16_kernel(const float* __restrict__ in,
                                unsigned short* __restrict__ out, int n) {
  int i = blockIdx.x * 256 + threadIdx.x;
  if (i < n) out[i] = f2bf(in[i]);
}

__global__ void fill_zero_us_kernel(unsigned short* __restrict__ p, int n) {
  int i = blockIdx.x * 256 + threadIdx.x;
  if (i < n) p[i] = 0;
}

// LayerNorm over last dim D, output bf16 with row remap:
// orow = (row / R) * obstride + ooff + (row % R)
__global__ void ln_bf16_kernel(const float* __restrict__ in,
                               const float* __restrict__ w,
                               const float* __restrict__ b,
                               unsigned short* __restrict__ out,
                               int D, int R, int obstride, int ooff) {
  int row = blockIdx.x;
  const float* x = in + (size_t)row * D;
  int orow = (row / R) * obstride + ooff + (row % R);
  unsigned short* y = out + (size_t)orow * D;

  __shared__ float red[256];
  float s = 0.f, ss = 0.f;
  for (int i = threadIdx.x; i < D; i += 256) { float v = x[i]; s += v; ss += v * v; }
  red[threadIdx.x] = s; __syncthreads();
  for (int st = 128; st > 0; st >>= 1) {
    if (threadIdx.x < st) red[threadIdx.x] += red[threadIdx.x + st];
    __syncthreads();
  }
  float mean = red[0] / D; __syncthreads();
  red[threadIdx.x] = ss; __syncthreads();
  for (int st = 128; st > 0; st >>= 1) {
    if (threadIdx.x < st) red[threadIdx.x] += red[threadIdx.x + st];
    __syncthreads();
  }
  float var = red[0] / D - mean * mean;
  float inv = rsqrtf(fmaxf(var, 0.f) + 1e-5f);
  for (int i = threadIdx.x; i < D; i += 256)
    y[i] = f2bf((x[i] - mean) * inv * w[i] + b[i]);
}

// split QKV [B*S, 3C] f32 -> q[B,H,N,HD] (scaled, +q_bias), k[B,H,S,HD], v^T[B,H,HD,SP]
__global__ void qkv_split_kernel(const float* __restrict__ qkv,
                                 const float* __restrict__ q_bias,
                                 const float* __restrict__ v_bias,
                                 unsigned short* __restrict__ Qo,
                                 unsigned short* __restrict__ Ko,
                                 unsigned short* __restrict__ Vto) {
  long long idx = (long long)blockIdx.x * 256 + threadIdx.x;
  const long long total = (long long)BB * SS * 3 * CC;
  if (idx >= total) return;
  int j = (int)(idx % (3 * CC));
  long long row = idx / (3 * CC);
  int s = (int)(row % SS);
  int b = (int)(row / SS);
  int part = j >> 10;
  int c = j & (CC - 1);
  int h = c >> 6, hd = c & 63;
  float v = qkv[idx];
  if (part == 0) {
    if (s >= II) {
      int n = s - II;
      Qo[(((size_t)(b * HH + h) * NN) + n) * HD + hd] =
          f2bf((v + q_bias[c]) * 0.125f);   // HD^-0.5 = 1/8
    }
  } else if (part == 1) {
    Ko[(((size_t)(b * HH + h) * SS) + s) * HD + hd] = f2bf(v);
  } else {
    Vto[(((size_t)(b * HH + h) * HD) + hd) * SP + s] = f2bf(v + v_bias[c]);
  }
}

// x1 = x + gamma1 * o_proj
__global__ void residual1_kernel(const float* __restrict__ x,
                                 const float* __restrict__ g1,
                                 const float* __restrict__ op,
                                 float* __restrict__ x1, int n) {
  int i = blockIdx.x * 256 + threadIdx.x;
  if (i < n) x1[i] = x[i] + g1[i & (CC - 1)] * op[i];
}

// out = x1 + gamma2*mlp + gate*pmlp
__global__ void final_kernel(const float* __restrict__ x1,
                             const float* __restrict__ g2,
                             const float* __restrict__ gate,
                             const float* __restrict__ mlp,
                             const float* __restrict__ pmlp,
                             float* __restrict__ out, int n) {
  int i = blockIdx.x * 256 + threadIdx.x;
  if (i < n) out[i] = x1[i] + g2[i & (CC - 1)] * mlp[i] + gate[0] * pmlp[i];
}

// ---------------- generic WMMA GEMM, 2x2 register blocking ----------------
// out[m,n] = act( sum_k A[m,k]*W[n,k] + bias[n] ), A/W bf16 row-major.
// 256 threads = 8 waves (2x4). Wave tile = 32(M) x 32(N) = 2x2 WMMA tiles,
// so each A/B fragment feeds two v_wmma ops. Block tile = 64(M) x 128(N).
// Store row remap: orow = (m/rin)*obstride + ooff + (m%rin)
template<bool GELU_, bool OUT_BF16>
__global__ __launch_bounds__(256)
void gemm_wmma_kernel(const unsigned short* __restrict__ A, int lda,
                      const unsigned short* __restrict__ W, int ldw,
                      const float* __restrict__ bias,
                      void* __restrict__ outp, int ldc,
                      int M, int N, int K,
                      int rin, int obstride, int ooff) {
  int lane = threadIdx.x & 31;
  int wave = threadIdx.x >> 5;
  int wm = wave & 1, wn = wave >> 1;          // 2 x 4 wave grid
  int m_base = blockIdx.x * 64 + wm * 32;
  int n_base = blockIdx.y * 128 + wn * 32;
  int lr = lane & 15, lg = lane >> 4;

  int arow0 = m_base + lr, arow1 = m_base + 16 + lr;
  bool av0 = (arow0 < M), av1 = (arow1 < M);
  const unsigned short* ap0 = A + (size_t)(av0 ? arow0 : 0) * lda + lg * 8;
  const unsigned short* ap1 = A + (size_t)(av1 ? arow1 : 0) * lda + lg * 8;
  int wcol0 = n_base + lr, wcol1 = n_base + 16 + lr;   // N % 128 == 0 here
  const unsigned short* wp0 = W + (size_t)wcol0 * ldw + lg * 8;
  const unsigned short* wp1 = W + (size_t)wcol1 * ldw + lg * 8;

  v8f acc00 = vzero8(), acc01 = vzero8(), acc10 = vzero8(), acc11 = vzero8();
  for (int k0 = 0; k0 < K; k0 += 32) {
    AFrag a0, a1, b0, b1;
    if (av0) { a0.u[0] = *(const uint4*)(ap0 + k0);
               a0.u[1] = *(const uint4*)(ap0 + k0 + 16); }
    else     { a0.u[0] = uz4(); a0.u[1] = uz4(); }
    if (av1) { a1.u[0] = *(const uint4*)(ap1 + k0);
               a1.u[1] = *(const uint4*)(ap1 + k0 + 16); }
    else     { a1.u[0] = uz4(); a1.u[1] = uz4(); }
    b0.u[0] = *(const uint4*)(wp0 + k0);
    b0.u[1] = *(const uint4*)(wp0 + k0 + 16);
    b1.u[0] = *(const uint4*)(wp1 + k0);
    b1.u[1] = *(const uint4*)(wp1 + k0 + 16);
    // stream-ahead prefetch (lowers to global_prefetch_b8; speculative, safe OOB)
    __builtin_prefetch(ap0 + k0 + 128, 0, 3);
    __builtin_prefetch(wp0 + k0 + 128, 0, 3);
    acc00 = wmma_bf16(a0, b0, acc00);
    acc01 = wmma_bf16(a0, b1, acc01);
    acc10 = wmma_bf16(a1, b0, acc10);
    acc11 = wmma_bf16(a1, b1, acc11);
  }

  float bv0 = bias ? bias[wcol0] : 0.f;
  float bv1 = bias ? bias[wcol1] : 0.f;
#pragma unroll
  for (int t = 0; t < 4; t++) {
    int mt = t >> 1, nt = t & 1;
    v8f acc = (t == 0) ? acc00 : (t == 1) ? acc01 : (t == 2) ? acc10 : acc11;
    float bv = nt ? bv1 : bv0;
    int col = nt ? wcol1 : wcol0;
#pragma unroll
    for (int i = 0; i < 8; i++) {
      int m = m_base + mt * 16 + 8 * lg + i;
      if (m < M) {
        float v = acc[i] + bv;
        if (GELU_) v = gelu_exact(v);
        int orow = (m / rin) * obstride + ooff + (m % rin);
        size_t oi = (size_t)orow * ldc + col;
        if (OUT_BF16) ((unsigned short*)outp)[oi] = f2bf(v);
        else          ((float*)outp)[oi] = v;
      }
    }
  }
}

// ---------------- flash attention ----------------
// One wave per (b, h, 16-query tile). Streams keys in chunks of 32 with
// online softmax. q[B,H,N,HD] pre-scaled bf16, k[B,H,S,HD] bf16,
// v^T[B,H,HD,SP] bf16 (padding zeroed). Writes o[B,N,C] bf16.
__global__ __launch_bounds__(32)
void flash_attn_kernel(const unsigned short* __restrict__ Q,
                       const unsigned short* __restrict__ Km,
                       const unsigned short* __restrict__ Vt,
                       unsigned short* __restrict__ O) {
  int bid = blockIdx.x;
  int qt = bid & 63;                 // N/16 = 64 tiles
  int h  = (bid >> 6) & (HH - 1);
  int b  = bid >> 10;
  int lane = threadIdx.x;
  int lr = lane & 15, lg = lane >> 4;

  const unsigned short* qbase =
      Q + (((size_t)(b * HH + h) * NN) + qt * 16 + lr) * HD + lg * 16;
  AFrag qa0, qa1;
  qa0.u[0] = *(const uint4*)(qbase);
  qa0.u[1] = *(const uint4*)(qbase + 8);
  qa1.u[0] = *(const uint4*)(qbase + 32);
  qa1.u[1] = *(const uint4*)(qbase + 40);

  const unsigned short* kbase = Km + (size_t)(b * HH + h) * SS * HD;
  const unsigned short* vbase = Vt + (size_t)(b * HH + h) * HD * SP;

  float mrow[8], lsum[8];
  v8f oacc[4];
#pragma unroll
  for (int j = 0; j < 4; j++) oacc[j] = vzero8();
#pragma unroll
  for (int i = 0; i < 8; i++) { mrow[i] = -3.0e38f; lsum[i] = 0.f; }

  __shared__ unsigned short plds[16 * 32];

  for (int c0 = 0; c0 < SS; c0 += 32) {
    // ---- scores: two 16x16 tiles over 32 keys, K-dim = HD = 2 wmma steps
    int key0 = c0 + lr, key1 = c0 + 16 + lr;
    AFrag b00, b01, b10, b11;
    const unsigned short* kp0 = kbase + (size_t)key0 * HD + lg * 16;
    const unsigned short* kp1 = kbase + (size_t)key1 * HD + lg * 16;
    if (key0 < SS) {
      b00.u[0] = *(const uint4*)(kp0);      b00.u[1] = *(const uint4*)(kp0 + 8);
      b01.u[0] = *(const uint4*)(kp0 + 32); b01.u[1] = *(const uint4*)(kp0 + 40);
    } else { b00.u[0]=uz4(); b00.u[1]=uz4(); b01.u[0]=uz4(); b01.u[1]=uz4(); }
    if (key1 < SS) {
      b10.u[0] = *(const uint4*)(kp1);      b10.u[1] = *(const uint4*)(kp1 + 8);
      b11.u[0] = *(const uint4*)(kp1 + 32); b11.u[1] = *(const uint4*)(kp1 + 40);
    } else { b10.u[0]=uz4(); b10.u[1]=uz4(); b11.u[0]=uz4(); b11.u[1]=uz4(); }

    v8f s0 = vzero8(), s1 = vzero8();
    s0 = wmma_bf16(qa0, b00, s0);
    s0 = wmma_bf16(qa1, b01, s0);
    s1 = wmma_bf16(qa0, b10, s1);
    s1 = wmma_bf16(qa1, b11, s1);

    // mask invalid key columns
    if (key0 >= SS) {
#pragma unroll
      for (int i = 0; i < 8; i++) s0[i] = -1.0e30f;
    }
    if (key1 >= SS) {
#pragma unroll
      for (int i = 0; i < 8; i++) s1[i] = -1.0e30f;
    }

    // ---- online softmax row stats (rows live across the 16 lanes of a half)
    float mc[8];
#pragma unroll
    for (int i = 0; i < 8; i++) mc[i] = fmaxf(s0[i], s1[i]);
    for (int d = 1; d < 16; d <<= 1) {
#pragma unroll
      for (int i = 0; i < 8; i++) mc[i] = fmaxf(mc[i], __shfl_xor(mc[i], d, 32));
    }
    float scale[8], rs[8];
#pragma unroll
    for (int i = 0; i < 8; i++) {
      float mn = fmaxf(mrow[i], mc[i]);
      scale[i] = __expf(mrow[i] - mn);
      mrow[i] = mn;
    }
#pragma unroll
    for (int i = 0; i < 8; i++) {
      s0[i] = __expf(s0[i] - mrow[i]);
      s1[i] = __expf(s1[i] - mrow[i]);
      rs[i] = s0[i] + s1[i];
    }
    for (int d = 1; d < 16; d <<= 1) {
#pragma unroll
      for (int i = 0; i < 8; i++) rs[i] += __shfl_xor(rs[i], d, 32);
    }
#pragma unroll
    for (int i = 0; i < 8; i++) lsum[i] = lsum[i] * scale[i] + rs[i];
#pragma unroll
    for (int j = 0; j < 4; j++)
#pragma unroll
      for (int i = 0; i < 8; i++) oacc[j][i] *= scale[i];

    // ---- C-layout P -> A-layout via LDS bounce
    __syncthreads();
#pragma unroll
    for (int i = 0; i < 8; i++) {
      int r = i + 8 * lg;
      plds[r * 32 + lr]      = f2bf(s0[i]);
      plds[r * 32 + 16 + lr] = f2bf(s1[i]);
    }
    __syncthreads();
    AFrag pf;
    pf.u[0] = *(const uint4*)&plds[lr * 32 + lg * 16];
    pf.u[1] = *(const uint4*)&plds[lr * 32 + lg * 16 + 8];

    // ---- P(16x32) @ V(32x64): 4 output tiles, V^T rows are contiguous keys
#pragma unroll
    for (int j = 0; j < 4; j++) {
      AFrag vf;
      const unsigned short* vp = vbase + (size_t)(j * 16 + lr) * SP + c0 + lg * 16;
      vf.u[0] = *(const uint4*)(vp);
      vf.u[1] = *(const uint4*)(vp + 8);
      oacc[j] = wmma_bf16(pf, vf, oacc[j]);
    }
  }

  // ---- epilogue: o[b, qrow, h*64 + hd] = O / l
#pragma unroll
  for (int j = 0; j < 4; j++) {
#pragma unroll
    for (int i = 0; i < 8; i++) {
      int qrow = qt * 16 + i + 8 * lg;
      int col = h * HD + j * 16 + lr;
      float v = oacc[j][i] / lsum[i];
      O[((size_t)(b * NN + qrow)) * CC + col] = f2bf(v);
    }
  }
}

// ---------------- host launch ----------------
extern "C" void kernel_launch(void* const* d_in, const int* in_sizes, int n_in,
                              void* d_out, int out_size, void* d_ws, size_t ws_size,
                              hipStream_t stream) {
  const float* x        = (const float*)d_in[0];
  const float* instr    = (const float*)d_in[1];
  const float* norm1_w  = (const float*)d_in[2];
  const float* norm1_b  = (const float*)d_in[3];
  const float* qkv_w    = (const float*)d_in[4];
  const float* q_bias   = (const float*)d_in[5];
  const float* v_bias   = (const float*)d_in[6];
  const float* proj_w   = (const float*)d_in[7];
  const float* proj_b   = (const float*)d_in[8];
  const float* gamma_1  = (const float*)d_in[9];
  const float* gamma_2  = (const float*)d_in[10];
  const float* norm2_w  = (const float*)d_in[11];
  const float* norm2_b  = (const float*)d_in[12];
  const float* fc1_w    = (const float*)d_in[13];
  const float* fc1_b    = (const float*)d_in[14];
  const float* fc2_w    = (const float*)d_in[15];
  const float* fc2_b    = (const float*)d_in[16];
  const float* pfc1_w   = (const float*)d_in[17];
  const float* pfc1_b   = (const float*)d_in[18];
  const float* pfc2_w   = (const float*)d_in[19];
  const float* pfc2_b   = (const float*)d_in[20];
  const float* pgate    = (const float*)d_in[21];
  const float* ir_ln_w  = (const float*)d_in[22];
  const float* ir_ln_b  = (const float*)d_in[23];
  const float* ir_l1_w  = (const float*)d_in[24];
  const float* ir_l1_b  = (const float*)d_in[25];
  const float* ir_l2_w  = (const float*)d_in[26];
  const float* ir_l2_b  = (const float*)d_in[27];
  float* out = (float*)d_out;
  (void)in_sizes; (void)n_in; (void)out_size; (void)ws_size;

  char* ws = (char*)d_ws;
  size_t off = 0;
  auto alloc = [&](size_t bytes) -> void* {
    void* p = ws + off;
    off = (off + bytes + 255) & ~(size_t)255;
    return p;
  };

  // bf16 weights
  unsigned short* qkv_wb  = (unsigned short*)alloc((size_t)3*CC*CC*2);
  unsigned short* proj_wb = (unsigned short*)alloc((size_t)CC*CC*2);
  unsigned short* fc1_wb  = (unsigned short*)alloc((size_t)HID*CC*2);
  unsigned short* fc2_wb  = (unsigned short*)alloc((size_t)CC*HID*2);
  unsigned short* pfc1_wb = (unsigned short*)alloc((size_t)HID*CC*2);
  unsigned short* pfc2_wb = (unsigned short*)alloc((size_t)CC*HID*2);
  unsigned short* irl1_wb = (unsigned short*)alloc((size_t)CC*IDD*2);
  unsigned short* irl2_wb = (unsigned short*)alloc((size_t)CC*CC*2);
  // activations
  unsigned short* lni  = (unsigned short*)alloc((size_t)BB*II*IDD*2);
  unsigned short* t1   = (unsigned short*)alloc((size_t)BB*II*CC*2);
  unsigned short* cat  = (unsigned short*)alloc((size_t)BB*SS*CC*2);
  float*          qkvt = (float*)alloc((size_t)BB*SS*3*CC*4);
  unsigned short* Qb   = (unsigned short*)alloc((size_t)BB*HH*NN*HD*2);
  unsigned short* Kb   = (unsigned short*)alloc((size_t)BB*HH*SS*HD*2);
  unsigned short* Vtb  = (unsigned short*)alloc((size_t)BB*HH*HD*SP*2);
  unsigned short* Ob   = (unsigned short*)alloc((size_t)BB*NN*CC*2);
  float*          oprj = (float*)alloc((size_t)BB*NN*CC*4);
  float*          x1   = (float*)alloc((size_t)BB*NN*CC*4);
  unsigned short* yb   = (unsigned short*)alloc((size_t)BB*NN*CC*2);
  unsigned short* h1   = (unsigned short*)alloc((size_t)BB*NN*HID*2);
  unsigned short* h2   = (unsigned short*)alloc((size_t)BB*NN*HID*2);
  float*          mlpf = (float*)alloc((size_t)BB*NN*CC*4);
  float*          pmlf = (float*)alloc((size_t)BB*NN*CC*4);

  auto cvt = [&](const float* src, unsigned short* dst, int n) {
    cvt_bf16_kernel<<<(n + 255) / 256, 256, 0, stream>>>(src, dst, n);
  };
  cvt(qkv_w,  qkv_wb,  3*CC*CC);
  cvt(proj_w, proj_wb, CC*CC);
  cvt(fc1_w,  fc1_wb,  HID*CC);
  cvt(fc2_w,  fc2_wb,  CC*HID);
  cvt(pfc1_w, pfc1_wb, HID*CC);
  cvt(pfc2_w, pfc2_wb, CC*HID);
  cvt(ir_l1_w, irl1_wb, CC*IDD);
  cvt(ir_l2_w, irl2_wb, CC*CC);

  {
    int n = BB*HH*HD*SP;
    fill_zero_us_kernel<<<(n + 255) / 256, 256, 0, stream>>>(Vtb, n);
  }

  // instruct LN -> lni
  ln_bf16_kernel<<<BB*II, 256, 0, stream>>>(instr, ir_ln_w, ir_ln_b, lni,
                                            IDD, BB*II, 0, 0);
  // t1 = gelu(lni @ irl1^T + b)   [308 x 1024]
  gemm_wmma_kernel<true,  true><<<dim3(5, 8), 256, 0, stream>>>(
      lni, IDD, irl1_wb, IDD, ir_l1_b, t1, CC, BB*II, CC, IDD, BB*II, 0, 0);
  // kv = t1 @ irl2^T + b -> scattered into cat rows [b*S + i]
  gemm_wmma_kernel<false, true><<<dim3(5, 8), 256, 0, stream>>>(
      t1, CC, irl2_wb, CC, ir_l2_b, cat, CC, BB*II, CC, CC, II, SS, 0);
  // norm1(x) -> cat rows [b*S + I + n]
  ln_bf16_kernel<<<BB*NN, 256, 0, stream>>>(x, norm1_w, norm1_b, cat,
                                            CC, NN, SS, II);
  // qkv = cat @ qkv_w^T   [4404 x 3072] f32
  gemm_wmma_kernel<false, false><<<dim3(69, 24), 256, 0, stream>>>(
      cat, CC, qkv_wb, CC, (const float*)nullptr, qkvt, 3*CC,
      BB*SS, 3*CC, CC, BB*SS, 0, 0);
  // split into per-head q (scaled+bias), k, v^T (bias)
  {
    long long n = (long long)BB*SS*3*CC;
    qkv_split_kernel<<<(int)((n + 255) / 256), 256, 0, stream>>>(
        qkvt, q_bias, v_bias, Qb, Kb, Vtb);
  }
  // flash attention -> Ob [B,N,C] bf16
  flash_attn_kernel<<<BB*HH*(NN/16), 32, 0, stream>>>(Qb, Kb, Vtb, Ob);
  // proj
  gemm_wmma_kernel<false, false><<<dim3(64, 8), 256, 0, stream>>>(
      Ob, CC, proj_wb, CC, proj_b, oprj, CC, BB*NN, CC, CC, BB*NN, 0, 0);
  // x1 = x + gamma1 * proj
  residual1_kernel<<<(BB*NN*CC + 255) / 256, 256, 0, stream>>>(
      x, gamma_1, oprj, x1, BB*NN*CC);
  // y = LN(x1)
  ln_bf16_kernel<<<BB*NN, 256, 0, stream>>>(x1, norm2_w, norm2_b, yb,
                                            CC, BB*NN, 0, 0);
  // mlp branch
  gemm_wmma_kernel<true,  true><<<dim3(64, 32), 256, 0, stream>>>(
      yb, CC, fc1_wb, CC, fc1_b, h1, HID, BB*NN, HID, CC, BB*NN, 0, 0);
  gemm_wmma_kernel<false, false><<<dim3(64, 8), 256, 0, stream>>>(
      h1, HID, fc2_wb, HID, fc2_b, mlpf, CC, BB*NN, CC, HID, BB*NN, 0, 0);
  // parallel mlp branch
  gemm_wmma_kernel<true,  true><<<dim3(64, 32), 256, 0, stream>>>(
      yb, CC, pfc1_wb, CC, pfc1_b, h2, HID, BB*NN, HID, CC, BB*NN, 0, 0);
  gemm_wmma_kernel<false, false><<<dim3(64, 8), 256, 0, stream>>>(
      h2, HID, pfc2_wb, HID, pfc2_b, pmlf, CC, BB*NN, CC, HID, BB*NN, 0, 0);
  // out = x1 + gamma2*mlp + gate*pmlp
  final_kernel<<<(BB*NN*CC + 255) / 256, 256, 0, stream>>>(
      x1, gamma_2, pgate, mlpf, pmlf, out, BB*NN*CC);
}
